// MambaCrossBlock_74371653697797
// MI455X (gfx1250) — matrix operand order repaired and checked by reading, not compiled, not run on hardware
//
// MI455X (gfx1250) Mamba + cross-attention + FFN block.
// All GEMMs (incl. attention QK^T / PV) via V_WMMA_F32_16X16X32_BF16, wave32.
// GEMM: 64x64 wave tiles (16 WMMA : 16 b128 loads per K-step), no edge guards
// (every GEMM N is padded to a multiple of the 256-wide block tile).
#include <hip/hip_runtime.h>
#include <hip/hip_bf16.h>
#include <stdint.h>
#include <stddef.h>

#define Bc   2
#define Lc   1024
#define Tc   512
#define Dc   1024
#define DIc  2048
#define DSc  16
#define DCc  4
#define DTRc 64
#define Hc   8
#define HDc  128     // D / H
#define PROJP 256    // x_proj output padded from 96 -> 256 (one block tile)

typedef __attribute__((ext_vector_type(16))) __bf16 bf16x16;
typedef __attribute__((ext_vector_type(8)))  float  floatx8;

__device__ __forceinline__ uint16_t f2bf(float x) {
  union { float f; uint32_t u; } c; c.f = x;
  return (uint16_t)((c.u + 0x7FFFu + ((c.u >> 16) & 1u)) >> 16);  // RNE
}

__device__ __forceinline__ floatx8 wmma_bf16(bf16x16 a, bf16x16 b, floatx8 c) {
  // (neg_a, A, neg_b, B, c_mod, C, reuse_a, reuse_b)
  return __builtin_amdgcn_wmma_f32_16x16x32_bf16(false, a, false, b, (short)0, c, false, false);
}

// ---------------------------------------------------------------------------
// Fragment packing: f32 -> bf16 in per-lane WMMA fragments, contiguous k order (same for A and B)
// A-frag (16x32, MxK): lane<16: M=lane,  K = kb + e (e=0..15), kb=kt*32
//                      lane>=16: M=lane-16, same with kb+=16
// B-frag (32x16, KxN): lane: N=lane&15, K = kt*32 + (lane>=16?16:0) + e
// dst layout: [tileM or tileN][tileK][lane][16]  (512 elems / fragment)
// ---------------------------------------------------------------------------
__global__ __launch_bounds__(256)
void pack_a_k(const float* __restrict__ src, uint16_t* __restrict__ dst,
              int M, int K, int ld) {
  int t = blockIdx.x * 256 + threadIdx.x;
  int KT = K >> 5;
  int total = (M >> 4) * KT * 32;
  if (t >= total) return;
  int lane = t & 31, tile = t >> 5;
  int kt = tile % KT, mt = tile / KT;
  int row = mt * 16 + (lane & 15);
  int kb  = kt * 32 + ((lane & 16) ? 16 : 0);
  const float* s = src + (size_t)row * ld + kb;
  uint32_t w[8];
#pragma unroll
  for (int e = 0; e < 4; ++e)
    w[e]     = (uint32_t)f2bf(s[2*e])      | ((uint32_t)f2bf(s[2*e+1])      << 16);
#pragma unroll
  for (int e = 0; e < 4; ++e)
    w[4 + e] = (uint32_t)f2bf(s[8 + 2*e]) | ((uint32_t)f2bf(s[8 + 2*e+1]) << 16);
  uint32_t* d = (uint32_t*)(dst + (size_t)t * 16);
#pragma unroll
  for (int e = 0; e < 8; ++e) d[e] = w[e];
}

// B from row-major W[K][N] (stride ld); packs Npad columns, zero-fills n>=Nsrc
__global__ __launch_bounds__(256)
void pack_b_k(const float* __restrict__ src, uint16_t* __restrict__ dst,
              int K, int Npad, int ld, int Nsrc) {
  int t = blockIdx.x * 256 + threadIdx.x;
  int KT = K >> 5;
  int total = (Npad >> 4) * KT * 32;
  if (t >= total) return;
  int lane = t & 31, tile = t >> 5;
  int kt = tile % KT, nt = tile / KT;
  int n   = nt * 16 + (lane & 15);
  int kb0 = kt * 32 + ((lane & 16) ? 16 : 0);
  uint32_t w[8];
  if (n < Nsrc) {
    const float* s = src + (size_t)kb0 * ld + n;
#pragma unroll
    for (int e = 0; e < 8; ++e)
      w[e] = (uint32_t)f2bf(s[(size_t)(2*e) * ld]) |
             ((uint32_t)f2bf(s[(size_t)(2*e + 1) * ld]) << 16);
  } else {
#pragma unroll
    for (int e = 0; e < 8; ++e) w[e] = 0u;
  }
  uint32_t* d = (uint32_t*)(dst + (size_t)t * 16);
#pragma unroll
  for (int e = 0; e < 8; ++e) d[e] = w[e];
}

// B from "transposed" storage src[n*ld + k] (e.g. K^T for attention): contiguous reads
__global__ __launch_bounds__(256)
void pack_bT_k(const float* __restrict__ src, uint16_t* __restrict__ dst,
               int K, int N, int ld) {
  int t = blockIdx.x * 256 + threadIdx.x;
  int KT = K >> 5;
  int total = (N >> 4) * KT * 32;
  if (t >= total) return;
  int lane = t & 31, tile = t >> 5;
  int kt = tile % KT, nt = tile / KT;
  int n   = nt * 16 + (lane & 15);
  int kb0 = kt * 32 + ((lane & 16) ? 16 : 0);
  const float* s = src + (size_t)n * ld + kb0;
  uint32_t w[8];
#pragma unroll
  for (int e = 0; e < 8; ++e)
    w[e] = (uint32_t)f2bf(s[2*e]) | ((uint32_t)f2bf(s[2*e + 1]) << 16);
  uint32_t* d = (uint32_t*)(dst + (size_t)t * 16);
#pragma unroll
  for (int e = 0; e < 8; ++e) d[e] = w[e];
}

// ---------------------------------------------------------------------------
// WMMA GEMM: C[M,N] = act(A@B + bias + res).
// 8 waves/block in 2x4; each wave owns a 64x64 tile (4x4 fragments):
// 16 v_wmma per K-step against 16 unconditional b128 loads.
// Block tile = 128x256; M % 128 == 0 and N % 256 == 0 guaranteed by caller.
// ACT: 0=none, 1=relu, 2=softplus.
// ---------------------------------------------------------------------------
template<int ACT, bool BIAS, bool RES>
__global__ __launch_bounds__(256)
void wmma_gemm(const uint16_t* __restrict__ Ap, const uint16_t* __restrict__ Bp,
               const float* __restrict__ bias, const float* __restrict__ res,
               float* __restrict__ C, int M, int N, int K) {
  const int KT = K >> 5;
  const int lane = threadIdx.x & 31;
  const int wave = threadIdx.x >> 5;
  const int mt0 = blockIdx.x * 8 + (wave >> 2) * 4;   // 4 M-tiles per wave
  const int nt0 = blockIdx.y * 16 + (wave & 3) * 4;   // 4 N-tiles per wave
  floatx8 acc[4][4] = {};
  const size_t lo = (size_t)lane * 16;
  const uint16_t* ap[4];
  const uint16_t* bp[4];
#pragma unroll
  for (int i = 0; i < 4; ++i) {
    ap[i] = Ap + (size_t)(mt0 + i) * KT * 512 + lo;
    bp[i] = Bp + (size_t)(nt0 + i) * KT * 512 + lo;
  }
  for (int kt = 0; kt < KT; ++kt) {
    const size_t ko = (size_t)kt * 512;
    bf16x16 a[4], b[4];
#pragma unroll
    for (int i = 0; i < 4; ++i) a[i] = *(const bf16x16*)(ap[i] + ko);
#pragma unroll
    for (int j = 0; j < 4; ++j) b[j] = *(const bf16x16*)(bp[j] + ko);
    __builtin_prefetch(ap[0] + ko + 512, 0, 1);
    __builtin_prefetch(bp[0] + ko + 512, 0, 1);
#pragma unroll
    for (int i = 0; i < 4; ++i)
#pragma unroll
      for (int j = 0; j < 4; ++j)
        acc[i][j] = wmma_bf16(a[i], b[j], acc[i][j]);
  }
  const int col = lane & 15;
  const int rbase = (lane >> 4) * 8;   // C layout: row = r + 8*(lane>=16)
#pragma unroll
  for (int i = 0; i < 4; ++i) {
    const int mt = mt0 + i;
#pragma unroll
    for (int j = 0; j < 4; ++j) {
      const int n = (nt0 + j) * 16 + col;
      const float bv = BIAS ? bias[n] : 0.0f;
#pragma unroll
      for (int r = 0; r < 8; ++r) {
        const int m = mt * 16 + rbase + r;
        float v = acc[i][j][r] + bv;
        if (RES) v += res[(size_t)m * N + n];
        if (ACT == 1) v = fmaxf(v, 0.0f);
        if (ACT == 2) v = (v > 20.0f) ? v : log1pf(__expf(v));
        C[(size_t)m * N + n] = v;
      }
    }
  }
}

// ---------------------------------------------------------------------------
// LayerNorm: one 256-thread block per row
// ---------------------------------------------------------------------------
__global__ __launch_bounds__(256)
void layernorm_k(const float* __restrict__ x, const float* __restrict__ w,
                 const float* __restrict__ b, float* __restrict__ y, int Dn) {
  const int row = blockIdx.x;
  const float* xr = x + (size_t)row * Dn;
  float s = 0.f, s2 = 0.f;
  for (int i = threadIdx.x; i < Dn; i += 256) { float v = xr[i]; s += v; s2 += v * v; }
#pragma unroll
  for (int o = 16; o > 0; o >>= 1) { s += __shfl_xor(s, o, 32); s2 += __shfl_xor(s2, o, 32); }
  __shared__ float sm[8], sm2[8];
  const int wave = threadIdx.x >> 5, lane = threadIdx.x & 31;
  if (lane == 0) { sm[wave] = s; sm2[wave] = s2; }
  __syncthreads();
  if (wave == 0) {
    s  = (lane < 8) ? sm[lane]  : 0.f;
    s2 = (lane < 8) ? sm2[lane] : 0.f;
#pragma unroll
    for (int o = 4; o > 0; o >>= 1) { s += __shfl_xor(s, o, 32); s2 += __shfl_xor(s2, o, 32); }
    if (lane == 0) { sm[0] = s; sm2[0] = s2; }
  }
  __syncthreads();
  const float mean = sm[0] / Dn;
  const float var  = sm2[0] / Dn - mean * mean;
  const float rs   = rsqrtf(var + 1e-5f);
  for (int i = threadIdx.x; i < Dn; i += 256)
    y[(size_t)row * Dn + i] = (xr[i] - mean) * rs * w[i] + b[i];
}

// ---------------------------------------------------------------------------
// Depthwise causal conv (DC=4) + SiLU on u = uz[:, :DI] (ld = 2*DI)
// ---------------------------------------------------------------------------
__global__ __launch_bounds__(256)
void conv_silu_k(const float* __restrict__ uz, const float* __restrict__ cw,
                 const float* __restrict__ cb, float* __restrict__ uc) {
  const size_t idx = (size_t)blockIdx.x * 256 + threadIdx.x;
  const int d = (int)(idx % DIc);
  const size_t row = idx / DIc;             // b*L + l
  const int l = (int)(row % Lc);
  float acc = cb[d];
#pragma unroll
  for (int j = 0; j < DCc; ++j) {
    const int ls = l - (DCc - 1) + j;
    if (ls >= 0) acc += uz[(row + (size_t)(ls - l)) * (2 * DIc) + d] * cw[d * DCc + j];
  }
  const float sig = 1.f / (1.f + __expf(-acc));
  uc[row * DIc + d] = acc * sig;
}

// ---------------------------------------------------------------------------
// Selective scan: one lane per (b, d) channel, DS=16 state in registers.
// dt already softplus'd; proj holds [dt(64) | B(16) | C(16) | pad] (ld PROJP).
// ---------------------------------------------------------------------------
__global__ __launch_bounds__(256)
void mamba_scan_k(const float* __restrict__ dt, const float* __restrict__ proj,
                  const float* __restrict__ u, const float* __restrict__ z,
                  const float* __restrict__ A_log, const float* __restrict__ Dp,
                  float* __restrict__ y) {
  const int d = blockIdx.x * 256 + threadIdx.x;
  const int b = blockIdx.y;
  if (d >= DIc) return;
  float Aa[DSc], hs[DSc];
#pragma unroll
  for (int s = 0; s < DSc; ++s) { Aa[s] = -__expf(A_log[d * DSc + s]); hs[s] = 0.f; }
  const float Dd = Dp[d];
  for (int l = 0; l < Lc; ++l) {
    const size_t row = (size_t)b * Lc + l;
    const float dtv = dt[row * DIc + d];
    const float uv  = u[row * DIc + d];
    const float zv  = z[row * (2 * DIc) + d];
    const float du  = dtv * uv;
    float yv = 0.f;
#pragma unroll
    for (int s = 0; s < DSc; ++s) {
      const float Bs = proj[row * PROJP + DTRc + s];
      const float Cs = proj[row * PROJP + DTRc + DSc + s];
      hs[s] = __expf(dtv * Aa[s]) * hs[s] + du * Bs;
      yv += hs[s] * Cs;
    }
    yv += uv * Dd;
    const float sig = 1.f / (1.f + __expf(-zv));
    y[row * DIc + d] = yv * (zv * sig);
  }
}

// ---------------------------------------------------------------------------
// Flash-style cross attention.  grid=(L/16/4, B*H), block=128 (4 waves).
// Wave owns one 16-row L tile: S=QK^T via WMMA, online softmax, P restaged
// through LDS (bf16) into A-fragment layout, O += P@V via WMMA.
// ---------------------------------------------------------------------------
__global__ __launch_bounds__(128)
void attn_k(const uint16_t* __restrict__ qA, const uint16_t* __restrict__ kB,
            const uint16_t* __restrict__ vB, const unsigned char* __restrict__ mask,
            float* __restrict__ out) {
  const int lane = threadIdx.x & 31;
  const int wave = threadIdx.x >> 5;
  const int lt = blockIdx.x * 4 + wave;           // L tile
  const int bh = blockIdx.y;
  const int b = bh >> 3, h = bh & 7;
  __shared__ __align__(16) uint16_t pbuf[4][16][32];
  const float scale = 0.0883883476483184f;        // 1/sqrt(128)

  bf16x16 qa[4];
  {
    const uint16_t* qp = qA + ((size_t)bh * 64 + lt) * 4 * 512 + (size_t)lane * 16;
#pragma unroll
    for (int kt = 0; kt < 4; ++kt) { qa[kt] = *(const bf16x16*)qp; qp += 512; }
  }
  floatx8 o[8] = {};
  float m8[8], l8[8];
#pragma unroll
  for (int r = 0; r < 8; ++r) { m8[r] = -INFINITY; l8[r] = 0.f; }

  const int col = lane & 15;
  const int rbase = (lane >> 4) * 8;

  for (int tp = 0; tp < Tc / 32; ++tp) {          // 32 T-columns per step
    floatx8 s0 = {}, s1 = {};
    const uint16_t* k0p = kB + ((size_t)bh * 32 + 2 * tp) * 4 * 512 + (size_t)lane * 16;
#pragma unroll
    for (int kt = 0; kt < 4; ++kt) {
      bf16x16 b0 = *(const bf16x16*)(k0p + (size_t)kt * 512);
      bf16x16 b1 = *(const bf16x16*)(k0p + (size_t)(4 + kt) * 512);
      s0 = wmma_bf16(qa[kt], b0, s0);
      s1 = wmma_bf16(qa[kt], b1, s1);
    }
    const int t0 = (2 * tp) * 16 + col;
    const bool v0 = mask[b * Tc + t0]      != 0;
    const bool v1 = mask[b * Tc + t0 + 16] != 0;
#pragma unroll
    for (int r = 0; r < 8; ++r) {
      s0[r] = v0 ? s0[r] * scale : -1e9f;
      s1[r] = v1 ? s1[r] * scale : -1e9f;
    }
#pragma unroll
    for (int r = 0; r < 8; ++r) {
      float mr = fmaxf(s0[r], s1[r]);
#pragma unroll
      for (int off = 1; off < 16; off <<= 1) mr = fmaxf(mr, __shfl_xor(mr, off, 32));
      const float mnew = fmaxf(m8[r], mr);
      const float corr = __expf(m8[r] - mnew);
      const float p0 = __expf(s0[r] - mnew);
      const float p1 = __expf(s1[r] - mnew);
      float ps = p0 + p1;
#pragma unroll
      for (int off = 1; off < 16; off <<= 1) ps += __shfl_xor(ps, off, 32);
      l8[r] = l8[r] * corr + ps;
      m8[r] = mnew;
#pragma unroll
      for (int j = 0; j < 8; ++j) o[j][r] *= corr;
      pbuf[wave][rbase + r][col]      = f2bf(p0);
      pbuf[wave][rbase + r][col + 16] = f2bf(p1);
    }
    // intra-wave LDS store->load ordering
    asm volatile("s_wait_dscnt 0" ::: "memory");
    union { uint4 q[2]; bf16x16 v; } pa;
    const int arow = lane & 15;
    const int kbo  = (lane >> 4) * 16;
    pa.q[0] = *(const uint4*)&pbuf[wave][arow][kbo];
    pa.q[1] = *(const uint4*)&pbuf[wave][arow][kbo + 8];
    const uint16_t* vp = vB + ((size_t)bh * 8 * 16 + tp) * 512 + (size_t)lane * 16;
#pragma unroll
    for (int j = 0; j < 8; ++j) {
      bf16x16 vb = *(const bf16x16*)(vp + (size_t)j * 16 * 512);
      o[j] = wmma_bf16(pa.v, vb, o[j]);
    }
  }
#pragma unroll
  for (int r = 0; r < 8; ++r) {
    const float inv = 1.0f / l8[r];
    const int m = lt * 16 + rbase + r;
    const size_t base = ((size_t)b * Lc + m) * Dc + h * HDc + col;
#pragma unroll
    for (int j = 0; j < 8; ++j) out[base + j * 16] = o[j][r] * inv;
  }
}

// ---------------------------------------------------------------------------
extern "C" void kernel_launch(void* const* d_in, const int* in_sizes, int n_in,
                              void* d_out, int out_size, void* d_ws, size_t ws_size,
                              hipStream_t stream) {
  (void)in_sizes; (void)n_in; (void)out_size; (void)ws_size;
  const float* x          = (const float*)d_in[0];
  const float* enc        = (const float*)d_in[1];
  const unsigned char* mask = (const unsigned char*)d_in[2];
  const float* n1w = (const float*)d_in[3],  *n1b = (const float*)d_in[4];
  const float* n2w = (const float*)d_in[5],  *n2b = (const float*)d_in[6];
  const float* cnw = (const float*)d_in[7],  *cnb = (const float*)d_in[8];
  const float* in_proj = (const float*)d_in[9];
  const float* conv_w  = (const float*)d_in[10], *conv_b = (const float*)d_in[11];
  const float* x_proj  = (const float*)d_in[12];
  const float* dt_proj = (const float*)d_in[13], *dt_b = (const float*)d_in[14];
  const float* A_log   = (const float*)d_in[15], *Dp   = (const float*)d_in[16];
  const float* out_proj = (const float*)d_in[17];
  const float* attn_in_w = (const float*)d_in[18], *attn_in_b = (const float*)d_in[19];
  const float* attn_out_w = (const float*)d_in[20], *attn_out_b = (const float*)d_in[21];
  const float* ffn_w1 = (const float*)d_in[22], *ffn_b1 = (const float*)d_in[23];
  const float* ffn_w2 = (const float*)d_in[24], *ffn_b2 = (const float*)d_in[25];

  char* ws = (char*)d_ws;
  size_t off = 0;
  auto alloc = [&](size_t bytes) -> void* {
    off = (off + 255) & ~(size_t)255;
    void* p = ws + off; off += bytes; return p;
  };
  const int ML = Bc * Lc;   // 2048
  const int MT = Bc * Tc;   // 1024
  float*    lnbuf = (float*)   alloc((size_t)ML * Dc * 4);
  uint16_t* Ap    = (uint16_t*)alloc((size_t)ML * 4096 * 2);
  uint16_t* Bp    = (uint16_t*)alloc((size_t)16 << 20);
  float*    uz    = (float*)   alloc((size_t)ML * 2 * DIc * 4);
  float*    uc    = (float*)   alloc((size_t)ML * DIc * 4);
  float*    proj  = (float*)   alloc((size_t)ML * PROJP * 4);
  float*    dtb   = (float*)   alloc((size_t)ML * DIc * 4);
  float*    yb    = (float*)   alloc((size_t)ML * DIc * 4);
  float*    hbuf  = (float*)   alloc((size_t)ML * Dc * 4);
  float*    qbuf  = (float*)   alloc((size_t)ML * Dc * 4);
  float*    kbuf  = (float*)   alloc((size_t)MT * Dc * 4);
  float*    vbuf  = (float*)   alloc((size_t)MT * Dc * 4);
  uint16_t* qAf   = (uint16_t*)alloc((size_t)16 * 64 * 4 * 512 * 2);
  uint16_t* kBf   = (uint16_t*)alloc((size_t)16 * 32 * 4 * 512 * 2);
  uint16_t* vBf   = (uint16_t*)alloc((size_t)16 * 8 * 16 * 512 * 2);
  float*    ao    = (float*)   alloc((size_t)ML * Dc * 4);
  float*    h2    = (float*)   alloc((size_t)ML * Dc * 4);
  float*    f1    = (float*)   alloc((size_t)ML * 4 * Dc * 4);

  auto blks = [](int total) { return dim3((unsigned)((total + 255) / 256)); };
  auto atot = [](int M, int K) { return (M / 16) * (K / 32) * 32; };
  auto btot = [](int K, int N) { return (N / 16) * (K / 32) * 32; };
  auto gg   = [](int M, int N) { return dim3((unsigned)(M / 128), (unsigned)(N / 256)); };

  // ---- Mamba branch ----
  layernorm_k<<<ML, 256, 0, stream>>>(x, n1w, n1b, lnbuf, Dc);
  pack_a_k<<<blks(atot(ML, Dc)), 256, 0, stream>>>(lnbuf, Ap, ML, Dc, Dc);
  pack_b_k<<<blks(btot(Dc, 2 * DIc)), 256, 0, stream>>>(in_proj, Bp, Dc, 2 * DIc, 2 * DIc, 2 * DIc);
  wmma_gemm<0, false, false><<<gg(ML, 2 * DIc), 256, 0, stream>>>(Ap, Bp, nullptr, nullptr, uz, ML, 2 * DIc, Dc);

  conv_silu_k<<<blks(ML * DIc), 256, 0, stream>>>(uz, conv_w, conv_b, uc);

  pack_a_k<<<blks(atot(ML, DIc)), 256, 0, stream>>>(uc, Ap, ML, DIc, DIc);
  pack_b_k<<<blks(btot(DIc, PROJP)), 256, 0, stream>>>(x_proj, Bp, DIc, PROJP, 96, 96);
  wmma_gemm<0, false, false><<<gg(ML, PROJP), 256, 0, stream>>>(Ap, Bp, nullptr, nullptr, proj, ML, PROJP, DIc);

  pack_a_k<<<blks(atot(ML, DTRc)), 256, 0, stream>>>(proj, Ap, ML, DTRc, PROJP);
  pack_b_k<<<blks(btot(DTRc, DIc)), 256, 0, stream>>>(dt_proj, Bp, DTRc, DIc, DIc, DIc);
  wmma_gemm<2, true, false><<<gg(ML, DIc), 256, 0, stream>>>(Ap, Bp, dt_b, nullptr, dtb, ML, DIc, DTRc);

  mamba_scan_k<<<dim3(DIc / 256, Bc), 256, 0, stream>>>(dtb, proj, uc, uz + DIc, A_log, Dp, yb);

  pack_a_k<<<blks(atot(ML, DIc)), 256, 0, stream>>>(yb, Ap, ML, DIc, DIc);
  pack_b_k<<<blks(btot(DIc, Dc)), 256, 0, stream>>>(out_proj, Bp, DIc, Dc, Dc, Dc);
  wmma_gemm<0, false, true><<<gg(ML, Dc), 256, 0, stream>>>(Ap, Bp, nullptr, x, hbuf, ML, Dc, DIc);

  // ---- Cross attention ----
  layernorm_k<<<ML, 256, 0, stream>>>(hbuf, cnw, cnb, lnbuf, Dc);
  pack_a_k<<<blks(atot(ML, Dc)), 256, 0, stream>>>(lnbuf, Ap, ML, Dc, Dc);
  const size_t bsz = (size_t)(Dc / 16) * (Dc / 32) * 512;
  pack_b_k<<<blks(btot(Dc, Dc)), 256, 0, stream>>>(attn_in_w,          Bp,           Dc, Dc, 3 * Dc, Dc);
  pack_b_k<<<blks(btot(Dc, Dc)), 256, 0, stream>>>(attn_in_w + Dc,     Bp + bsz,     Dc, Dc, 3 * Dc, Dc);
  pack_b_k<<<blks(btot(Dc, Dc)), 256, 0, stream>>>(attn_in_w + 2 * Dc, Bp + 2 * bsz, Dc, Dc, 3 * Dc, Dc);
  wmma_gemm<0, true, false><<<gg(ML, Dc), 256, 0, stream>>>(Ap, Bp, attn_in_b, nullptr, qbuf, ML, Dc, Dc);
  pack_a_k<<<blks(atot(MT, Dc)), 256, 0, stream>>>(enc, Ap, MT, Dc, Dc);
  wmma_gemm<0, true, false><<<gg(MT, Dc), 256, 0, stream>>>(Ap, Bp + bsz,     attn_in_b + Dc,     nullptr, kbuf, MT, Dc, Dc);
  wmma_gemm<0, true, false><<<gg(MT, Dc), 256, 0, stream>>>(Ap, Bp + 2 * bsz, attn_in_b + 2 * Dc, nullptr, vbuf, MT, Dc, Dc);

  for (int bh = 0; bh < Bc * Hc; ++bh) {
    const int b = bh >> 3, h = bh & 7;
    pack_a_k<<<blks(atot(Lc, HDc)), 256, 0, stream>>>(
        qbuf + (size_t)b * Lc * Dc + h * HDc, qAf + (size_t)bh * 64 * 4 * 512, Lc, HDc, Dc);
    pack_bT_k<<<blks(btot(HDc, Tc)), 256, 0, stream>>>(
        kbuf + (size_t)b * Tc * Dc + h * HDc, kBf + (size_t)bh * 32 * 4 * 512, HDc, Tc, Dc);
    pack_b_k<<<blks(btot(Tc, HDc)), 256, 0, stream>>>(
        vbuf + (size_t)b * Tc * Dc + h * HDc, vBf + (size_t)bh * 8 * 16 * 512, Tc, HDc, Dc, HDc);
  }
  attn_k<<<dim3(Lc / 16 / 4, Bc * Hc), 128, 0, stream>>>(qAf, kBf, vBf, mask, ao);

  pack_a_k<<<blks(atot(ML, Dc)), 256, 0, stream>>>(ao, Ap, ML, Dc, Dc);
  pack_b_k<<<blks(btot(Dc, Dc)), 256, 0, stream>>>(attn_out_w, Bp, Dc, Dc, Dc, Dc);
  wmma_gemm<0, true, true><<<gg(ML, Dc), 256, 0, stream>>>(Ap, Bp, attn_out_b, hbuf, h2, ML, Dc, Dc);

  // ---- FFN ----
  layernorm_k<<<ML, 256, 0, stream>>>(h2, n2w, n2b, lnbuf, Dc);
  pack_a_k<<<blks(atot(ML, Dc)), 256, 0, stream>>>(lnbuf, Ap, ML, Dc, Dc);
  pack_b_k<<<blks(btot(Dc, 4 * Dc)), 256, 0, stream>>>(ffn_w1, Bp, Dc, 4 * Dc, 4 * Dc, 4 * Dc);
  wmma_gemm<1, true, false><<<gg(ML, 4 * Dc), 256, 0, stream>>>(Ap, Bp, ffn_b1, nullptr, f1, ML, 4 * Dc, Dc);
  pack_a_k<<<blks(atot(ML, 4 * Dc)), 256, 0, stream>>>(f1, Ap, ML, 4 * Dc, 4 * Dc);
  pack_b_k<<<blks(btot(4 * Dc, Dc)), 256, 0, stream>>>(ffn_w2, Bp, 4 * Dc, Dc, Dc, Dc);
  wmma_gemm<0, true, true><<<gg(ML, Dc), 256, 0, stream>>>(Ap, Bp, ffn_b2, h2, (float*)d_out, ML, Dc, 4 * Dc);
}
